// GATrEncoder_83099027243677
// MI455X (gfx1250) — matrix operand order repaired and checked by reading, not compiled, not run on hardware
//
#include <hip/hip_runtime.h>
#include <hip/hip_bf16.h>
#include <stdint.h>

typedef _Float16 f16;
typedef __attribute__((ext_vector_type(16))) _Float16 v16h;
typedef __attribute__((ext_vector_type(8)))  _Float16 v8h;
typedef __attribute__((ext_vector_type(8)))  float    v8f;

#define BT 8
#define NT 1024
#define MTOK (BT*NT)     // 8192 tokens
#define CH 128           // channels
#define NB 16            // blade components
#define HEADS 8

// ---------------- Cl(3,0,1) blade tables (compile-time, fully constant-folded) ----
struct Tabs {
  signed char gp_s[16][16];
  signed char gp_k[16][16];
  signed char op_s[16][16];
};
constexpr int BM[16] = {0,1,2,4,8,3,5,6,9,10,12,7,11,13,14,15}; // idx -> bitmask
constexpr int idx_of(int bm){ for (int i=0;i<16;i++) if (BM[i]==bm) return i; return 0; }
constexpr int pc4(int x){ int c=0; for(int i=0;i<4;i++) c+=(x>>i)&1; return c; }
constexpr int rsign(int a,int b){ int s=0; int t=a>>1; while(t){ s+=pc4(t&b); t>>=1;} return (s&1)?-1:1; }
constexpr Tabs mk_tabs(){
  Tabs t{};
  for(int i=0;i<16;i++) for(int j=0;j<16;j++){
    int a=BM[i], b=BM[j];
    t.gp_k[i][j]=(signed char)idx_of(a^b);
    if (a&b&1){ t.gp_s[i][j]=0; t.op_s[i][j]=0; }
    else { int s=rsign(a,b); t.gp_s[i][j]=(signed char)s; t.op_s[i][j]=((a&b)==0)?(signed char)s:0; }
  }
  return t;
}
constexpr Tabs TAB = mk_tabs();
constexpr int GRADE_OF[16]={0,1,1,1,1,2,2,2,2,2,2,3,3,3,3,4};
constexpr int NDIDX[8]={0,2,3,4,7,9,10,14};

// ---------------- WMMA fragment helpers (wave32, 16x16x32 f16) --------------------
// A fragment 16x32: lane L holds row (L&15); elems 0..7 = K kb..kb+7, 8..15 = K kb+16..kb+23, kb = (L>=16)?8:0
__device__ __forceinline__ v16h frag_a(const f16* base, int stride, int k0){
  int lane = threadIdx.x & 31;
  const f16* p = base + (size_t)(lane & 15)*stride + k0 + ((lane >> 4) << 3);
  union { v16h v; v8h h[2]; } u;
  u.h[0] = *(const v8h*)p;
  u.h[1] = *(const v8h*)(p + 16);
  return u.v;
}
// B fragment 32x16: lane L holds col (L&15); elems 0..7 = K kb..kb+7, 8..15 = K kb+16..kb+23, kb = (L>=16)?8:0
// "base" rows are the N dimension (row-major, K contiguous)
__device__ __forceinline__ v16h frag_b(const f16* base, int stride, int k0){
  int lane = threadIdx.x & 31;
  const f16* p = base + (size_t)(lane & 15)*stride + k0 + ((lane >> 4) << 3);
  union { v16h v; v8h h[2]; } u;
  u.h[0] = *(const v8h*)p;
  u.h[1] = *(const v8h*)(p + 16);
  return u.v;
}
__device__ __forceinline__ v8f wmma16(v16h a, v16h b, v8f c){
  return __builtin_amdgcn_wmma_f32_16x16x32_f16(false, a, false, b, (short)0, c, false, false);
}

__device__ __forceinline__ float geluf(float x){
  float x3 = x*x*x;
  return 0.5f*x*(1.0f + tanhf(0.7978845608028654f*(x + 0.044715f*x3)));
}

// ---------------- embed: rays + points -> equi_linear(w_in) -> X fp32 planar ------
__global__ void embed_kernel(const float* __restrict__ pts, const float* __restrict__ view,
                             const float* __restrict__ tx, const float* __restrict__ w_in,
                             float* __restrict__ X){
  int idx = blockIdx.x*blockDim.x + threadIdx.x;
  if (idx >= MTOK*CH) return;
  int m = idx >> 7, o = idx & 127;
  float px=pts[3*m], py=pts[3*m+1], pz=pts[3*m+2];
  float vx=view[3*m], vy=view[3*m+1], vz=view[3*m+2];
  float qx=tx[3*m], qy=tx[3*m+1], qz=tx[3*m+2];
  float nrm = sqrtf(vx*vx+vy*vy+vz*vz);
  float inv = 1.0f/(nrm+1e-9f);
  float dx=vx*inv, dy=vy*inv, dz=vz*inv;
  float mx=py*dz-pz*dy, my=pz*dx-px*dz, mz=px*dy-py*dx;
  float ray[16]={0.f}; float pt[16]={0.f};
  ray[10]=dx; ray[9]=dy; ray[7]=dz; ray[5]=mx; ray[6]=my; ray[8]=mz;   // e23,e13,e12,e01,e02,e03
  pt[11]=qx; pt[12]=qy; pt[13]=qz; pt[14]=1.0f;                        // e012,e013,e023,e123
  #pragma unroll
  for (int k=0;k<16;k++){
    int g = GRADE_OF[k];
    float w0 = w_in[(g*CH+o)*2+0];
    float w1 = w_in[(g*CH+o)*2+1];
    X[((size_t)k*MTOK + m)*CH + o] = ray[k]*w0 + pt[k]*w1;
  }
}

// ---------------- equi layer norm (fp32 planar -> f16 planar) ---------------------
__global__ void norm_kernel(const float* __restrict__ X, f16* __restrict__ D){
  int m = blockIdx.x;
  int c = threadIdx.x; // 128 threads
  float sq = 0.f;
  #pragma unroll
  for (int j=0;j<8;j++){
    float v = X[((size_t)NDIDX[j]*MTOK + m)*CH + c];
    sq += v*v;
  }
  __shared__ float red[128];
  red[c]=sq; __syncthreads();
  for (int s=64;s>0;s>>=1){ if (c<s) red[c]+=red[c+s]; __syncthreads(); }
  float factor = rsqrtf(red[0]*(1.0f/128.0f) + 1e-5f);
  #pragma unroll
  for (int k=0;k<16;k++){
    size_t o = ((size_t)k*MTOK + m)*CH + c;
    D[o] = (f16)(X[o]*factor);
  }
}

// ---------------- generic f32 -> f16 cast and residual add ------------------------
__global__ void cast_kernel(const float* __restrict__ S, f16* __restrict__ D, int n){
  int i = blockIdx.x*blockDim.x + threadIdx.x;
  if (i < n) D[i] = (f16)S[i];
}
__global__ void add_kernel(float* __restrict__ X, const float* __restrict__ T, int n){
  int i = blockIdx.x*blockDim.x + threadIdx.x;
  if (i < n) X[i] += T[i];
}

// ---------------- WMMA equi_linear GEMM -------------------------------------------
// A: f16 planar [NBLADES][MTOK][128]; W: f16 [5][COUT][128]; Out: planar [NBLADES][MTOK][COUT]
// All 8 waves of a block share one blade (512 m-tiles/blade is a multiple of 8),
// so the grade's 16/32KB weight plane is staged once in LDS per block.
template<int COUT, int NBLADES, typename OUTT>
__global__ void equi_gemm_kernel(const f16* __restrict__ A, const f16* __restrict__ W,
                                 OUTT* __restrict__ Out){
  __shared__ __align__(16) f16 WS[COUT*128];
  const int mtiles = MTOK/16;
  // block-uniform blade / grade
  int kblk = (blockIdx.x*(int)(blockDim.x>>5)) / mtiles;
  {
    const v8h* src = (const v8h*)(W + (size_t)GRADE_OF[kblk]*COUT*128);
    v8h* dst = (v8h*)WS;
    for (int t = threadIdx.x; t < COUT*128/8; t += blockDim.x) dst[t] = src[t];
  }
  __syncthreads();

  int wg = blockIdx.x*(blockDim.x>>5) + (threadIdx.x>>5);
  int k  = wg / mtiles;           // == kblk (uniform), kept for address math clarity
  int mt = wg % mtiles;
  const f16* Ab = A + ((size_t)k*MTOK + mt*16)*128;
  OUTT* Ob = Out + ((size_t)k*MTOK + mt*16)*COUT;
  v16h a0 = frag_a(Ab,128,0), a1 = frag_a(Ab,128,32), a2 = frag_a(Ab,128,64), a3 = frag_a(Ab,128,96);
  int lane = threadIdx.x & 31;
  int cl = lane & 15, rb = (lane>>4)*8;
  #pragma unroll
  for (int nt=0; nt<COUT/16; ++nt){
    v8f acc = {0.f,0.f,0.f,0.f,0.f,0.f,0.f,0.f};
    const f16* Wn = WS + (size_t)nt*16*128;
    acc = wmma16(a0, frag_b(Wn,128,0),  acc);
    acc = wmma16(a1, frag_b(Wn,128,32), acc);
    acc = wmma16(a2, frag_b(Wn,128,64), acc);
    acc = wmma16(a3, frag_b(Wn,128,96), acc);
    #pragma unroll
    for (int r=0;r<8;r++) Ob[(size_t)(rb+r)*COUT + nt*16 + cl] = (OUTT)acc[r];
  }
}

// ---------------- attention packing (f16 planar in) -------------------------------
// Qh/Kh: [b][h][n][f] f16, f = c_local*8 + nondegen_j  (scale folded into Q)
__global__ void pack_qk_kernel(const f16* __restrict__ Qf, f16* __restrict__ Qh, float scale){
  int idx = blockIdx.x*blockDim.x + threadIdx.x;
  if (idx >= MTOK*HEADS*128) return;
  int f  = idx & 127;
  int n  = (idx >> 7) & (NT-1);
  int bh = idx >> 17;
  int b = bh >> 3, hh = bh & 7;
  int clocal = f >> 3, j = f & 7;
  int c = hh*16 + clocal;
  int m = b*NT + n;
  Qh[idx] = (f16)((float)Qf[((size_t)NDIDX[j]*MTOK + m)*CH + c] * scale);
}
// Vt: [b][h][g][n] f16 (feature-major!), g = c_local*16 + comp
__global__ void pack_v_kernel(const f16* __restrict__ Vf, f16* __restrict__ Vt){
  int idx = blockIdx.x*blockDim.x + threadIdx.x;
  if (idx >= MTOK*HEADS*256) return;
  int n  = idx & (NT-1);
  int g  = (idx >> 10) & 255;
  int bh = idx >> 18;
  int b = bh >> 3, hh = bh & 7;
  int clocal = g >> 4, kcomp = g & 15;
  int c = hh*16 + clocal;
  int m = b*NT + n;
  Vt[idx] = Vf[((size_t)kcomp*MTOK + m)*CH + c];
}

// ---------------- attention: 2 waves/block, 32 queries, full 1024-key row in LDS ---
// Output written directly as f16 planar [16][MTOK][128] (feeds the attn_out GEMM).
__global__ void attention_kernel(const f16* __restrict__ Qh, const f16* __restrict__ Kh,
                                 const f16* __restrict__ Vt, f16* __restrict__ AO){
  extern __shared__ char smem[];
  float* Slog = (float*)smem;               // [32][1024] fp32 logits
  f16*   Pm   = (f16*)(smem + 32*1024*4);   // [32][1024] f16 probabilities

  int bid = blockIdx.x;
  int qt  = bid & 31;          // 32 query tiles of 32
  int bh  = bid >> 5;          // b*8+h
  int b = bh >> 3, hh = bh & 7;
  int wave = threadIdx.x >> 5;
  int lane = threadIdx.x & 31;
  int cl = lane & 15, rb = (lane>>4)*8;

  int q0 = qt*32 + wave*16;    // this wave's 16 query rows within (b,h)
  const f16* Qb = Qh + ((size_t)bh*NT + q0)*128;
  v16h qa0 = frag_a(Qb,128,0), qa1 = frag_a(Qb,128,32), qa2 = frag_a(Qb,128,64), qa3 = frag_a(Qb,128,96);

  // Phase 1: logits = Q K^T (scale already folded into Q)
  const f16* Kb0 = Kh + (size_t)bh*NT*128;
  for (int kt=0; kt<NT/16; ++kt){
    const f16* Kb = Kb0 + (size_t)kt*16*128;
    v8f acc = {0.f,0.f,0.f,0.f,0.f,0.f,0.f,0.f};
    acc = wmma16(qa0, frag_b(Kb,128,0),  acc);
    acc = wmma16(qa1, frag_b(Kb,128,32), acc);
    acc = wmma16(qa2, frag_b(Kb,128,64), acc);
    acc = wmma16(qa3, frag_b(Kb,128,96), acc);
    #pragma unroll
    for (int r=0;r<8;r++) Slog[(wave*16+rb+r)*1024 + kt*16 + cl] = acc[r];
  }
  __syncthreads();

  // Phase 2: row softmax (wave 0's 32 lanes each own one of the 32 rows)
  if (threadIdx.x < 32){
    float* row = Slog + threadIdx.x*1024;
    f16*  prow = Pm   + threadIdx.x*1024;
    float mx = -1e30f;
    for (int i=0;i<1024;i++) mx = fmaxf(mx, row[i]);
    float s = 0.f;
    for (int i=0;i<1024;i++){ float e = __expf(row[i]-mx); s += e; row[i] = e; }
    float invs = 1.0f/s;
    for (int i=0;i<1024;i++) prow[i] = (f16)(row[i]*invs);
  }
  __syncthreads();

  // Phase 3: O = P @ V  (V is feature-major so B fragments are contiguous)
  const f16* Pw = Pm + (size_t)wave*16*1024;
  const f16* Vb = Vt + (size_t)bh*256*NT;
  for (int ft=0; ft<16; ++ft){            // 16 feature tiles of 16 (256 features)
    v8f acc = {0.f,0.f,0.f,0.f,0.f,0.f,0.f,0.f};
    const f16* Vrow = Vb + (size_t)ft*16*NT;
    for (int ks=0; ks<NT/32; ++ks){
      acc = wmma16(frag_a(Pw,1024,ks*32), frag_b(Vrow,NT,ks*32), acc);
    }
    // col cl = blade comp, ft = channel-in-head; write f16 planar [comp][m][c]
    #pragma unroll
    for (int r=0;r<8;r++){
      int q = q0 + rb + r;
      AO[((size_t)cl*MTOK + (b*NT + q))*CH + hh*16 + ft] = (f16)acc[r];
    }
  }
}

// ---------------- geometric product / outer product + scalar-gated gelu -----------
// inputs f16 planar [16][MTOK][64]; output f16 planar [16][MTOK][128] (gp | join)
__global__ void gp_gate_kernel(const f16* __restrict__ GPL, const f16* __restrict__ GPR,
                               const f16* __restrict__ JL,  const f16* __restrict__ JR,
                               f16* __restrict__ Out){
  int idx = blockIdx.x*blockDim.x + threadIdx.x;
  if (idx >= MTOK*64) return;
  int c = idx & 63, m = idx >> 6;
  float a[16], bb[16], r[16];
  #pragma unroll
  for (int k=0;k<16;k++){
    a[k]  = (float)GPL[((size_t)k*MTOK + m)*64 + c];
    bb[k] = (float)GPR[((size_t)k*MTOK + m)*64 + c];
    r[k] = 0.f;
  }
  #pragma unroll
  for (int i=0;i<16;i++)
    #pragma unroll
    for (int j=0;j<16;j++){
      int s = TAB.gp_s[i][j];
      if (s) r[TAB.gp_k[i][j]] += (float)s * a[i] * bb[j];
    }
  float gate = geluf(r[0]);
  #pragma unroll
  for (int k=0;k<16;k++) Out[((size_t)k*MTOK + m)*CH + c] = (f16)(r[k]*gate);

  #pragma unroll
  for (int k=0;k<16;k++){
    a[k]  = (float)JL[((size_t)k*MTOK + m)*64 + c];
    bb[k] = (float)JR[((size_t)k*MTOK + m)*64 + c];
    r[k] = 0.f;
  }
  #pragma unroll
  for (int i=0;i<16;i++)
    #pragma unroll
    for (int j=0;j<16;j++){
      int s = TAB.op_s[i][j];
      if (s) r[TAB.gp_k[i][j]] += (float)s * a[i] * bb[j];
    }
  gate = geluf(r[0]);
  #pragma unroll
  for (int k=0;k<16;k++) Out[((size_t)k*MTOK + m)*CH + 64 + c] = (f16)(r[k]*gate);
}

// ---------------- workspace layout -------------------------------------------------
constexpr size_t SZ_PLANE32 = (size_t)NB*MTOK*CH*sizeof(float);   // 64 MB
constexpr size_t SZ_PLANE16 = (size_t)NB*MTOK*CH*sizeof(f16);     // 32 MB
constexpr size_t SZ_HALF16  = (size_t)NB*MTOK*64*sizeof(f16);     // 16 MB
constexpr size_t SZ_QH = (size_t)BT*HEADS*NT*128*sizeof(f16);     // 16 MB
constexpr size_t SZ_VT = (size_t)BT*HEADS*NT*256*sizeof(f16);     // 32 MB
constexpr size_t OFF_X    = 0;
constexpr size_t OFF_TMP  = OFF_X + SZ_PLANE32;        // fp32 GEMM out (residual adds)
constexpr size_t OFF_F16  = OFF_TMP + SZ_PLANE32;      // f16 GEMM input staging
constexpr size_t OFF_F16B = OFF_F16 + SZ_PLANE16;      // f16 GEMM output staging
constexpr size_t OFF_QH   = OFF_F16B + SZ_PLANE16;
constexpr size_t OFF_KH   = OFF_QH + SZ_QH;
constexpr size_t OFF_VT   = OFF_KH + SZ_QH;            // QH+KH+VT = 64 MB
constexpr size_t OFF_P4   = OFF_QH;                    // overlay: QKV dead after attention
constexpr size_t OFF_WH   = OFF_P4 + 4*SZ_HALF16;      // == end of VT region
constexpr size_t NW_FULL = (size_t)5*128*128; // 81920
constexpr size_t NW_HALF = (size_t)5*64*128;  // 40960
constexpr size_t WS_NEED = OFF_WH + (7*NW_FULL + 4*NW_HALF)*sizeof(f16);

extern "C" void kernel_launch(void* const* d_in, const int* in_sizes, int n_in,
                              void* d_out, int out_size, void* d_ws, size_t ws_size,
                              hipStream_t stream){
  (void)in_sizes; (void)n_in; (void)out_size;
  if (ws_size < WS_NEED) return;

  const float* pts   = (const float*)d_in[0];
  const float* view  = (const float*)d_in[1];
  const float* tx    = (const float*)d_in[2];
  const float* w_in  = (const float*)d_in[3];
  const float* w_q   = (const float*)d_in[4];
  const float* w_k   = (const float*)d_in[5];
  const float* w_v   = (const float*)d_in[6];
  const float* w_ao  = (const float*)d_in[7];
  const float* w_mi  = (const float*)d_in[8];
  const float* w_gpl = (const float*)d_in[9];
  const float* w_gpr = (const float*)d_in[10];
  const float* w_jl  = (const float*)d_in[11];
  const float* w_jr  = (const float*)d_in[12];
  const float* w_mo  = (const float*)d_in[13];
  const float* w_out = (const float*)d_in[14];

  char* ws = (char*)d_ws;
  float* X   = (float*)(ws + OFF_X);
  float* TMP = (float*)(ws + OFF_TMP);
  f16*  F16  = (f16*)(ws + OFF_F16);
  f16*  F16B = (f16*)(ws + OFF_F16B);
  f16*  Qh   = (f16*)(ws + OFF_QH);
  f16*  Kh   = (f16*)(ws + OFF_KH);
  f16*  Vt   = (f16*)(ws + OFF_VT);
  f16*  P4   = (f16*)(ws + OFF_P4);
  f16*  WH   = (f16*)(ws + OFF_WH);

  f16* WHq   = WH;
  f16* WHk   = WHq  + NW_FULL;
  f16* WHv   = WHk  + NW_FULL;
  f16* WHao  = WHv  + NW_FULL;
  f16* WHmi  = WHao + NW_FULL;
  f16* WHmo  = WHmi + NW_FULL;
  f16* WHout = WHmo + NW_FULL;
  f16* WHgpl = WHout+ NW_FULL;
  f16* WHgpr = WHgpl+ NW_HALF;
  f16* WHjl  = WHgpr+ NW_HALF;
  f16* WHjr  = WHjl + NW_HALF;

  auto gr = [](size_t n){ return dim3((unsigned)((n + 255)/256)); };
  const int TB = 256;
  const int NPLANE = NB*MTOK*CH;
  const size_t QSTR = (size_t)NB*MTOK*64; // f16 elements per half-width plane set

  // pack weights to f16
  cast_kernel<<<gr(NW_FULL),TB,0,stream>>>(w_q,  WHq,  (int)NW_FULL);
  cast_kernel<<<gr(NW_FULL),TB,0,stream>>>(w_k,  WHk,  (int)NW_FULL);
  cast_kernel<<<gr(NW_FULL),TB,0,stream>>>(w_v,  WHv,  (int)NW_FULL);
  cast_kernel<<<gr(NW_FULL),TB,0,stream>>>(w_ao, WHao, (int)NW_FULL);
  cast_kernel<<<gr(NW_FULL),TB,0,stream>>>(w_mi, WHmi, (int)NW_FULL);
  cast_kernel<<<gr(NW_FULL),TB,0,stream>>>(w_mo, WHmo, (int)NW_FULL);
  cast_kernel<<<gr(NW_FULL),TB,0,stream>>>(w_out,WHout,(int)NW_FULL);
  cast_kernel<<<gr(NW_HALF),TB,0,stream>>>(w_gpl,WHgpl,(int)NW_HALF);
  cast_kernel<<<gr(NW_HALF),TB,0,stream>>>(w_gpr,WHgpr,(int)NW_HALF);
  cast_kernel<<<gr(NW_HALF),TB,0,stream>>>(w_jl, WHjl, (int)NW_HALF);
  cast_kernel<<<gr(NW_HALF),TB,0,stream>>>(w_jr, WHjr, (int)NW_HALF);

  // embed + input projection
  embed_kernel<<<gr((size_t)MTOK*CH),TB,0,stream>>>(pts,view,tx,w_in,X);

  // --- attention sub-block ---
  norm_kernel<<<MTOK,128,0,stream>>>(X,F16);
  equi_gemm_kernel<128,16,f16><<<1024,256,0,stream>>>(F16,WHq,F16B);
  pack_qk_kernel<<<gr((size_t)MTOK*HEADS*128),TB,0,stream>>>(F16B,Qh,0.08838834764831845f); // 1/sqrt(128)
  equi_gemm_kernel<128,16,f16><<<1024,256,0,stream>>>(F16,WHk,F16B);
  pack_qk_kernel<<<gr((size_t)MTOK*HEADS*128),TB,0,stream>>>(F16B,Kh,1.0f);
  equi_gemm_kernel<128,16,f16><<<1024,256,0,stream>>>(F16,WHv,F16B);
  pack_v_kernel<<<gr((size_t)MTOK*HEADS*256),TB,0,stream>>>(F16B,Vt);

  // F16 (normed x) is dead now -> reuse as attention output
  attention_kernel<<<BT*HEADS*(NT/32),64,32*1024*4 + 32*1024*2,stream>>>(Qh,Kh,Vt,F16);

  equi_gemm_kernel<128,16,float><<<1024,256,0,stream>>>(F16,WHao,TMP);
  add_kernel<<<gr((size_t)NPLANE),TB,0,stream>>>(X,TMP,NPLANE);

  // --- geometric MLP sub-block ---
  norm_kernel<<<MTOK,128,0,stream>>>(X,F16);
  equi_gemm_kernel<128,16,f16><<<1024,256,0,stream>>>(F16,WHmi,F16B);
  equi_gemm_kernel<64,16,f16><<<1024,256,0,stream>>>(F16B,WHgpl,P4 + 0*QSTR);
  equi_gemm_kernel<64,16,f16><<<1024,256,0,stream>>>(F16B,WHgpr,P4 + 1*QSTR);
  equi_gemm_kernel<64,16,f16><<<1024,256,0,stream>>>(F16B,WHjl, P4 + 2*QSTR);
  equi_gemm_kernel<64,16,f16><<<1024,256,0,stream>>>(F16B,WHjr, P4 + 3*QSTR);
  gp_gate_kernel<<<gr((size_t)MTOK*64),TB,0,stream>>>(P4 + 0*QSTR, P4 + 1*QSTR,
                                                      P4 + 2*QSTR, P4 + 3*QSTR, F16);
  equi_gemm_kernel<128,16,float><<<1024,256,0,stream>>>(F16,WHmo,TMP);
  add_kernel<<<gr((size_t)NPLANE),TB,0,stream>>>(X,TMP,NPLANE);

  // --- output projection: only scalar blade (component 0) is needed ---
  cast_kernel<<<gr((size_t)MTOK*CH),TB,0,stream>>>(X,F16,MTOK*CH); // blade-0 plane
  equi_gemm_kernel<128,1,float><<<64,256,0,stream>>>(F16,WHout,(float*)d_out);
}
